// BitLinear_81449759801592
// MI455X (gfx1250) — matrix operand order repaired and checked by reading, not compiled, not run on hardware
//
#include <hip/hip_runtime.h>

typedef __attribute__((ext_vector_type(16))) _Float16 v16h;
typedef __attribute__((ext_vector_type(8)))  _Float16 v8h;
typedef __attribute__((ext_vector_type(4)))  _Float16 v4h;
typedef __attribute__((ext_vector_type(8)))  float    v8f;
typedef __attribute__((ext_vector_type(4)))  int      v4i;

union frag_u { v16h v; struct { v8h lo; v8h hi; } h; };

#define QMAX 127.0f
#define BM 128
#define BN 128
#define BK 32
#define LDSK (BK + 8)   // 40 halves = 80 B row stride (16B aligned, staggers banks)

// CDNA5 async global->LDS path (GLOBAL_LOAD_ASYNC_TO_LDS_B128, ASYNCcnt).
// Probe result (round 2 diagnostic): param0 = v4i addrspace(1)*, param1 = v4i addrspace(3)*,
// then imm offset, imm cpol.
#if defined(__HIP_DEVICE_COMPILE__) && \
    __has_builtin(__builtin_amdgcn_global_load_async_to_lds_b128) && \
    __has_builtin(__builtin_amdgcn_s_wait_asynccnt)
#define USE_ASYNC_LDS 1
typedef __attribute__((address_space(1))) v4i* g_v4i_ptr;
typedef __attribute__((address_space(3))) v4i* l_v4i_ptr;
#define ASYNC_CP_B128(g, l)                                                  \
    __builtin_amdgcn_global_load_async_to_lds_b128((g_v4i_ptr)(g),           \
                                                   (l_v4i_ptr)(l), 0, 0)
#else
#define USE_ASYNC_LDS 0
#endif

// ---------------------------------------------------------------------------
// Pre-pass 1: fused GradualActQuant -> f16 Aq
// ---------------------------------------------------------------------------
__global__ void quant_act_kernel(const float* __restrict__ x,
                                 const float* __restrict__ rmax,
                                 _Float16* __restrict__ Aq,
                                 int total4) {
    int i = blockIdx.x * blockDim.x + threadIdx.x;
    if (i >= total4) return;
    const float s   = rmax[0] * (1.0f / QMAX);
    const float inv = 1.0f / (s + 1e-6f);
    const float4 v = ((const float4*)x)[i];
    float r0 = 0.5f * v.x + 0.5f * (fminf(fmaxf(rintf(v.x * inv), -QMAX), QMAX) * s);
    float r1 = 0.5f * v.y + 0.5f * (fminf(fmaxf(rintf(v.y * inv), -QMAX), QMAX) * s);
    float r2 = 0.5f * v.z + 0.5f * (fminf(fmaxf(rintf(v.z * inv), -QMAX), QMAX) * s);
    float r3 = 0.5f * v.w + 0.5f * (fminf(fmaxf(rintf(v.w * inv), -QMAX), QMAX) * s);
    v4h o = { (_Float16)r0, (_Float16)r1, (_Float16)r2, (_Float16)r3 };
    ((v4h*)Aq)[i] = o;
}

// ---------------------------------------------------------------------------
// Pre-pass 2: fused weight binarize + sprinkle + blend -> f16 Bq
// ---------------------------------------------------------------------------
__global__ void quant_wgt_kernel(const float* __restrict__ w,
                                 const unsigned char* __restrict__ mask,
                                 const float* __restrict__ pbs,
                                 _Float16* __restrict__ Bq,
                                 int in_div4, int total4) {
    int i = blockIdx.x * blockDim.x + threadIdx.x;
    if (i >= total4) return;
    const int row = i / in_div4;
    const float scale = pbs[row];
    const float4 v = ((const float4*)w)[i];
    const unsigned int mb = ((const unsigned int*)mask)[i];
    float wh0 = (v.x >= 0.0f) ? scale : -scale;
    float wh1 = (v.y >= 0.0f) ? scale : -scale;
    float wh2 = (v.z >= 0.0f) ? scale : -scale;
    float wh3 = (v.w >= 0.0f) ? scale : -scale;
    float r0 = (mb & 0x000000FFu) ? wh0 : 0.5f * (v.x + wh0);
    float r1 = (mb & 0x0000FF00u) ? wh1 : 0.5f * (v.y + wh1);
    float r2 = (mb & 0x00FF0000u) ? wh2 : 0.5f * (v.z + wh2);
    float r3 = (mb & 0xFF000000u) ? wh3 : 0.5f * (v.w + wh3);
    v4h o = { (_Float16)r0, (_Float16)r1, (_Float16)r2, (_Float16)r3 };
    ((v4h*)Bq)[i] = o;
}

// ---------------------------------------------------------------------------
// Double-buffered WMMA GEMM: out[n][o] = (Σ_k Aq[n][k]*Bq[o][k])*fscale[o]+fbias[o]
//   128x128 C-tile / 8 wave32; 64x32 per wave = 4x2 wmma accumulators.
//   Slab s+1 streams into buffer q (async global->LDS) while buffer p feeds WMMA.
// ---------------------------------------------------------------------------
__global__ __launch_bounds__(256)
void bitlinear_gemm_kernel(const _Float16* __restrict__ Aq,   // [M][K]
                           const _Float16* __restrict__ Bq,   // [N][K]
                           const float* __restrict__ fscale,
                           const float* __restrict__ fbias,
                           float* __restrict__ out,
                           int M, int Ndim, int K) {
    __shared__ _Float16 As[2][BM * LDSK];
    __shared__ _Float16 Bs[2][BN * LDSK];

    const int tid   = threadIdx.x;
    const int lane  = tid & 31;
    const int wave  = tid >> 5;
    const int waveM = wave >> 2;             // 0..1
    const int waveN = wave & 3;              // 0..3
    const int blockMBase = blockIdx.y * BM;
    const int blockNBase = blockIdx.x * BN;
    const int laneLo  = lane & 15;
    const int laneSel = lane >> 4;

    // 512 16B chunks per 128x32 tile -> 2 chunks per thread per tile
    const int c0 = tid, c1 = tid + 256;
    const int r0 = c0 >> 2, col0 = (c0 & 3) << 3;
    const int r1 = c1 >> 2, col1 = (c1 & 3) << 3;

    const _Float16* gA0 = Aq + (size_t)(blockMBase + r0) * K + col0;
    const _Float16* gA1 = Aq + (size_t)(blockMBase + r1) * K + col1;
    const _Float16* gB0 = Bq + (size_t)(blockNBase + r0) * K + col0;
    const _Float16* gB1 = Bq + (size_t)(blockNBase + r1) * K + col1;
    const int lA0 = r0 * LDSK + col0, lA1 = r1 * LDSK + col1;

    v8f acc[4][2];
#pragma unroll
    for (int ms = 0; ms < 4; ++ms)
#pragma unroll
        for (int ns = 0; ns < 2; ++ns)
            acc[ms][ns] = (v8f){0.f, 0.f, 0.f, 0.f, 0.f, 0.f, 0.f, 0.f};

    const int aRow0 = waveM * 64;
    const int bRow0 = waveN * 32;
    const int nStages = K / BK;

    // ---- prologue: slab 0 -> buffer 0 ----
#if USE_ASYNC_LDS
    ASYNC_CP_B128(gA0, &As[0][lA0]);
    ASYNC_CP_B128(gA1, &As[0][lA1]);
    ASYNC_CP_B128(gB0, &Bs[0][lA0]);
    ASYNC_CP_B128(gB1, &Bs[0][lA1]);
    __builtin_amdgcn_s_wait_asynccnt(0);
#else
    *(v8h*)(&As[0][lA0]) = *(const v8h*)gA0;
    *(v8h*)(&As[0][lA1]) = *(const v8h*)gA1;
    *(v8h*)(&Bs[0][lA0]) = *(const v8h*)gB0;
    *(v8h*)(&Bs[0][lA1]) = *(const v8h*)gB1;
#endif
    __syncthreads();

    for (int s = 0; s < nStages; ++s) {
        const int p = s & 1;
        const int q = p ^ 1;
        const bool hasNext = (s + 1) < nStages;
        const int kNext = (s + 1) * BK;

        // ---- issue loads for slab s+1 into buffer q (overlap with compute) ----
#if USE_ASYNC_LDS
        if (hasNext) {
            ASYNC_CP_B128(gA0 + kNext, &As[q][lA0]);
            ASYNC_CP_B128(gA1 + kNext, &As[q][lA1]);
            ASYNC_CP_B128(gB0 + kNext, &Bs[q][lA0]);
            ASYNC_CP_B128(gB1 + kNext, &Bs[q][lA1]);
        }
#else
        v8h na0, na1, nb0, nb1;
        if (hasNext) {
            na0 = *(const v8h*)(gA0 + kNext);
            na1 = *(const v8h*)(gA1 + kNext);
            nb0 = *(const v8h*)(gB0 + kNext);
            nb1 = *(const v8h*)(gB1 + kNext);
            if (s + 2 < nStages) {           // L2 prefetch two slabs ahead
                __builtin_prefetch(gA0 + kNext + BK, 0, 1);
                __builtin_prefetch(gB0 + kNext + BK, 0, 1);
            }
        }
#endif

        // ---- compute one K=32 WMMA step from buffer p ----
        v16h afrag[4], bfrag[2];
        // A frag (16x32): lanes 0-15 -> K 0..7 & 16..23, lanes 16-31 -> K 8..15 & 24..31
#pragma unroll
        for (int ms = 0; ms < 4; ++ms) {
            const _Float16* pa = &As[p][(aRow0 + ms * 16 + laneLo) * LDSK + laneSel * 8];
            frag_u u;
            u.h.lo = *(const v8h*)(pa);
            u.h.hi = *(const v8h*)(pa + 16);
            afrag[ms] = u.v;
        }
        // B frag (32x16): lane n = column n; lanes 0-15 -> K 0..7 & 16..23, lanes 16-31 -> K 8..15 & 24..31
#pragma unroll
        for (int ns = 0; ns < 2; ++ns) {
            const _Float16* pb = &Bs[p][(bRow0 + ns * 16 + laneLo) * LDSK + laneSel * 8];
            frag_u u;
            u.h.lo = *(const v8h*)(pb);
            u.h.hi = *(const v8h*)(pb + 16);
            bfrag[ns] = u.v;
        }
#pragma unroll
        for (int ms = 0; ms < 4; ++ms)
#pragma unroll
            for (int ns = 0; ns < 2; ++ns)
                acc[ms][ns] = __builtin_amdgcn_wmma_f32_16x16x32_f16(
                    false, afrag[ms], false, bfrag[ns],
                    (short)0, acc[ms][ns], false, false);

        // ---- commit slab s+1 into buffer q, single barrier per stage ----
#if USE_ASYNC_LDS
        if (hasNext) __builtin_amdgcn_s_wait_asynccnt(0);
#else
        if (hasNext) {
            *(v8h*)(&As[q][lA0]) = na0;
            *(v8h*)(&As[q][lA1]) = na1;
            *(v8h*)(&Bs[q][lA0]) = nb0;
            *(v8h*)(&Bs[q][lA1]) = nb1;
        }
#endif
        __syncthreads();
    }

    // ---- epilogue: scale/bias; C layout: VGPR r -> rows r / r+8, lane -> col ----
#pragma unroll
    for (int ns = 0; ns < 2; ++ns) {
        const int col = blockNBase + bRow0 + ns * 16 + laneLo;
        const float sc = fscale[col];
        const float bi = fbias[col];
#pragma unroll
        for (int ms = 0; ms < 4; ++ms) {
            const int rowBase = blockMBase + aRow0 + ms * 16 + laneSel * 8;
#pragma unroll
            for (int r = 0; r < 8; ++r) {
                out[(size_t)(rowBase + r) * Ndim + col] = acc[ms][ns][r] * sc + bi;
            }
        }
    }
}

// ---------------------------------------------------------------------------
extern "C" void kernel_launch(void* const* d_in, const int* in_sizes, int n_in,
                              void* d_out, int out_size, void* d_ws, size_t ws_size,
                              hipStream_t stream) {
    const float*         x    = (const float*)d_in[0];
    const float*         w    = (const float*)d_in[1];
    const float*         pbs  = (const float*)d_in[2];
    const float*         fsc  = (const float*)d_in[3];
    const float*         fbi  = (const float*)d_in[4];
    const float*         rmax = (const float*)d_in[5];
    const unsigned char* mask = (const unsigned char*)d_in[6];
    float*               out  = (float*)d_out;

    const int OUT = in_sizes[3];              // 2048
    const int IN  = in_sizes[1] / OUT;        // 2048
    const int M   = in_sizes[0] / IN;         // 8192

    _Float16* Aq = (_Float16*)d_ws;                 // M*IN   f16 (32 MB)
    _Float16* Bq = Aq + (size_t)M * IN;             // OUT*IN f16 ( 8 MB)

    const int totalA4 = (M * IN) / 4;
    const int totalW4 = (OUT * IN) / 4;

    quant_act_kernel<<<(totalA4 + 255) / 256, 256, 0, stream>>>(x, rmax, Aq, totalA4);
    quant_wgt_kernel<<<(totalW4 + 255) / 256, 256, 0, stream>>>(w, mask, pbs, Bq,
                                                                IN / 4, totalW4);

    dim3 grid(OUT / BN, M / BM);   // 16 x 64 workgroups
    bitlinear_gemm_kernel<<<grid, 256, 0, stream>>>(Aq, Bq, fsc, fbi, out, M, OUT, IN);
}
